// Attention_17755394802171
// MI455X (gfx1250) — matrix operand order repaired and checked by reading, not compiled, not run on hardware
//
#include <hip/hip_runtime.h>

// ---------------------------------------------------------------------------
// GPT-2 attention block on gfx1250 (CDNA5, wave32) using v_wmma_f32_16x16x32_bf16.
// Pipeline: [qkv GEMM + bias + scale + head-split]  ->  [flash attention]
//           -> [output projection GEMM + bias].
// All matmuls run on the WMMA pipe with bf16 operands / f32 accumulation.
// Pure-copy LDS staging uses GLOBAL_LOAD_ASYNC_TO_LDS_B128 (ASYNCcnt) when the
// toolchain exposes the builtin; otherwise falls back to VGPR staging.
// ---------------------------------------------------------------------------

typedef __bf16 bf16;
typedef __attribute__((ext_vector_type(16))) __bf16 v16bf;
typedef __attribute__((ext_vector_type(8)))  float  v8f;
typedef __attribute__((ext_vector_type(4)))  unsigned int u32x4;

union FragBF { v16bf v; u32x4 u[2]; };

__device__ __forceinline__ v8f wmma_bf16(v16bf a, v16bf b, v8f c) {
    // (neg_a, A, neg_b, B, c_mod, C, reuse_a, reuse_b)
    return __builtin_amdgcn_wmma_f32_16x16x32_bf16(false, a, false, b, (short)0, c,
                                                   false, false);
}

// ---- async memory->LDS copy (CDNA5 GLOBAL_LOAD_ASYNC_TO_LDS_B128) ----------
#if defined(__has_builtin)
#if __has_builtin(__builtin_amdgcn_global_load_async_to_lds_b128)
#define HAS_ASYNC_LDS 1
#endif
#endif
#ifndef HAS_ASYNC_LDS
#define HAS_ASYNC_LDS 0
#endif

// The builtin's params are pointers to int32x4 (vector_size(16)) in
// addrspace(1) (global) and addrspace(3) (LDS), per the compiler diagnostic.
typedef int v4i __attribute__((vector_size(16)));
typedef __attribute__((address_space(1))) v4i* g_v4i;
typedef __attribute__((address_space(3))) v4i* l_v4i;

// copy 16 bytes from global to LDS (per-lane addresses), async when available
__device__ __forceinline__ void async_copy16(const void* gsrc, void* ldst) {
#if HAS_ASYNC_LDS
    __builtin_amdgcn_global_load_async_to_lds_b128((g_v4i)gsrc, (l_v4i)ldst, 0, 0);
#else
    *(u32x4*)ldst = *(const u32x4*)gsrc;
#endif
}

__device__ __forceinline__ void wait_async() {
#if HAS_ASYNC_LDS
#if defined(__has_builtin) && __has_builtin(__builtin_amdgcn_s_wait_asynccnt)
    __builtin_amdgcn_s_wait_asynccnt(0);
#else
    asm volatile("s_wait_asynccnt 0x0" ::: "memory");
#endif
#endif
}

constexpr int Bc  = 4;      // batch
constexpr int S   = 2048;   // sequence
constexpr int D   = 1024;   // model dim
constexpr int H   = 16;     // heads
constexpr int HD  = 64;     // head dim
constexpr int D3  = 3072;   // 3*D

// ---------------------------------------------------------------------------
// Kernel 1: qkv = x @ w_attn + b_attn  (M=8192, K=1024, N=3072)
// Block tile 128x128, 256 threads = 8 waves; wave grid 4(M) x 2(N),
// each wave: 2x4 grid of 16x16 WMMA tiles, K stepped by 32.
// Epilogue: q scaled by 1/8, scatter q/v -> [B,H,S,64] bf16, k -> [B,H,64,S].
// (staging converts fp32->bf16, so it keeps the VGPR path)
// ---------------------------------------------------------------------------
__global__ __launch_bounds__(256) void k_gemm_qkv(const float* __restrict__ x,
                                                  const float* __restrict__ w,
                                                  const float* __restrict__ bias,
                                                  bf16* __restrict__ qws,
                                                  bf16* __restrict__ ktws,
                                                  bf16* __restrict__ vws) {
    __shared__ __align__(16) bf16 AL[128][32];
    __shared__ __align__(16) bf16 BL[32][128];

    const int t     = threadIdx.x;
    const int lane  = t & 31;
    const int wv    = t >> 5;
    const int waveM = wv & 3;        // 0..3  -> 32-row slabs
    const int waveN = wv >> 2;       // 0..1  -> 64-col slabs
    const int m0    = blockIdx.y * 128;
    const int n0    = blockIdx.x * 128;

    v8f acc[2][4];
#pragma unroll
    for (int mt = 0; mt < 2; ++mt)
#pragma unroll
        for (int nt = 0; nt < 4; ++nt)
            acc[mt][nt] = (v8f){0.f,0.f,0.f,0.f,0.f,0.f,0.f,0.f};

    const int rA = lane & 15;        // A-frag row within 16
    const int hA = lane >> 4;        // A-frag K-half selector

    for (int k0 = 0; k0 < D; k0 += 32) {
        // ---- stage A tile (128x32 f32 -> bf16) ----
        {
            const int r  = t >> 1;
            const int cb = (t & 1) * 16;
            const float4* src = (const float4*)(x + (size_t)(m0 + r) * D + k0 + cb);
            bf16 tmp[16];
#pragma unroll
            for (int j = 0; j < 4; ++j) {
                float4 f = src[j];
                tmp[4*j+0] = (bf16)f.x; tmp[4*j+1] = (bf16)f.y;
                tmp[4*j+2] = (bf16)f.z; tmp[4*j+3] = (bf16)f.w;
            }
            *(u32x4*)&AL[r][cb]     = ((u32x4*)tmp)[0];
            *(u32x4*)&AL[r][cb + 8] = ((u32x4*)tmp)[1];
        }
        // ---- stage B tile (32x128 f32 -> bf16) ----
        {
            const int r  = t >> 3;
            const int cb = (t & 7) * 16;
            const float4* src = (const float4*)(w + (size_t)(k0 + r) * D3 + n0 + cb);
            bf16 tmp[16];
#pragma unroll
            for (int j = 0; j < 4; ++j) {
                float4 f = src[j];
                tmp[4*j+0] = (bf16)f.x; tmp[4*j+1] = (bf16)f.y;
                tmp[4*j+2] = (bf16)f.z; tmp[4*j+3] = (bf16)f.w;
            }
            *(u32x4*)&BL[r][cb]     = ((u32x4*)tmp)[0];
            *(u32x4*)&BL[r][cb + 8] = ((u32x4*)tmp)[1];
        }
        __syncthreads();

        FragBF af[2], bf[4];
#pragma unroll
        for (int mt = 0; mt < 2; ++mt) {
            const int row = waveM * 32 + mt * 16 + rA;
            af[mt].u[0] = *(const u32x4*)&AL[row][hA * 8];
            af[mt].u[1] = *(const u32x4*)&AL[row][16 + hA * 8];
        }
#pragma unroll
        for (int nt = 0; nt < 4; ++nt) {
            const int col = waveN * 64 + nt * 16;
            for (int s = 0; s < 8; ++s) {
                bf[nt].v[s]     = BL[hA * 8 + s][col + rA];
                bf[nt].v[8 + s] = BL[16 + hA * 8 + s][col + rA];
            }
        }
#pragma unroll
        for (int mt = 0; mt < 2; ++mt)
#pragma unroll
            for (int nt = 0; nt < 4; ++nt)
                acc[mt][nt] = wmma_bf16(af[mt].v, bf[nt].v, acc[mt][nt]);
        __syncthreads();
    }

    // ---- epilogue: bias, q-scale, head-split scatter ----
    const int cN   = lane & 15;
    const int hsel = lane >> 4;
#pragma unroll
    for (int mt = 0; mt < 2; ++mt)
#pragma unroll
        for (int nt = 0; nt < 4; ++nt)
#pragma unroll
            for (int i = 0; i < 8; ++i) {
                const int m = m0 + waveM * 32 + mt * 16 + i + hsel * 8;
                const int n = n0 + waveN * 64 + nt * 16 + cN;
                float v = acc[mt][nt][i] + bias[n];
                const int sec = n >> 10;            // 0=q 1=k 2=v
                const int d   = n & 1023;
                const int h   = d >> 6;
                const int di  = d & 63;
                const int bb  = m >> 11;
                const int s   = m & 2047;
                const int bh  = bb * H + h;
                if (sec == 0) {
                    v *= 0.125f;                     // fold 1/sqrt(hd) into q
                    qws[((size_t)bh * S + s) * HD + di] = (bf16)v;
                } else if (sec == 1) {
                    ktws[((size_t)bh * HD + di) * S + s] = (bf16)v;   // transposed
                } else {
                    vws[((size_t)bh * S + s) * HD + di] = (bf16)v;
                }
            }
}

// ---------------------------------------------------------------------------
// Kernel 2: causal flash attention per (b,h) / 128-query tile.
// 8 waves; wave w owns query rows [w*16, w*16+16). Key tiles of 64.
// Q/K^T/V tiles are staged with async memory->LDS copies.
// ---------------------------------------------------------------------------
__global__ __launch_bounds__(256) void k_flash(const bf16* __restrict__ qws,
                                               const bf16* __restrict__ ktws,
                                               const bf16* __restrict__ vws,
                                               bf16* __restrict__ aws) {
    __shared__ __align__(16) bf16  QL[128][64];
    __shared__ __align__(16) bf16  KTL[64][64];   // [hd][key]
    __shared__ __align__(16) bf16  VL[64][64];    // [key][hd]
    __shared__ __align__(16) float SL[128][64];
    __shared__ __align__(16) bf16  PL[128][64];
    __shared__ float mL[128], lL[128], alphaL[128];

    const int t    = threadIdx.x;
    const int lane = t & 31;
    const int wv   = t >> 5;
    const int qt   = blockIdx.x;
    const int bh   = blockIdx.y;
    const int bb   = bh >> 4;
    const int h    = bh & 15;

    // stage Q tile (async copy, 64B per thread)
    {
        const int r  = t >> 1;
        const int cb = (t & 1) * 32;
        const bf16* src = qws + ((size_t)bh * S + qt * 128 + r) * HD + cb;
        bf16* dst = &QL[r][cb];
#pragma unroll
        for (int j = 0; j < 4; ++j)
            async_copy16(src + 8 * j, dst + 8 * j);
    }
    if (t < 128) { mL[t] = -1e30f; lL[t] = 0.f; }

    v8f O[4];
#pragma unroll
    for (int nt = 0; nt < 4; ++nt)
        O[nt] = (v8f){0.f,0.f,0.f,0.f,0.f,0.f,0.f,0.f};

    const int rA   = lane & 15;
    const int hA   = lane >> 4;
    const int nk   = qt * 2 + 2;     // causal: key tiles covering [0, qt*128+128)

    for (int kt = 0; kt < nk; ++kt) {
        const int kb = kt * 64;
        __syncthreads();             // protect KTL/VL/PL from previous iteration
        // stage K^T and V tiles (async copies, 32B each per thread)
        {
            const int r  = t >> 2;
            const int cb = (t & 3) * 16;
            const bf16* sk = ktws + ((size_t)bh * HD + r) * S + kb + cb;
            async_copy16(sk,     &KTL[r][cb]);
            async_copy16(sk + 8, &KTL[r][cb + 8]);
            const bf16* sv = vws + ((size_t)bh * S + kb + r) * HD + cb;
            async_copy16(sv,     &VL[r][cb]);
            async_copy16(sv + 8, &VL[r][cb + 8]);
        }
        wait_async();                // ASYNCcnt -> 0 before publishing tiles
        __syncthreads();

        // S = Q @ K^T (16x64 per wave, K(=hd) stepped by 32)
        v8f sc[4];
#pragma unroll
        for (int nt = 0; nt < 4; ++nt)
            sc[nt] = (v8f){0.f,0.f,0.f,0.f,0.f,0.f,0.f,0.f};
#pragma unroll
        for (int kk = 0; kk < 64; kk += 32) {
            FragBF a;
            a.u[0] = *(const u32x4*)&QL[wv * 16 + rA][kk + hA * 8];
            a.u[1] = *(const u32x4*)&QL[wv * 16 + rA][kk + 16 + hA * 8];
#pragma unroll
            for (int nt = 0; nt < 4; ++nt) {
                FragBF b;
                for (int s = 0; s < 8; ++s) {
                    b.v[s]     = KTL[kk + hA * 8 + s][nt * 16 + rA];
                    b.v[8 + s] = KTL[kk + 16 + hA * 8 + s][nt * 16 + rA];
                }
                sc[nt] = wmma_bf16(a.v, b.v, sc[nt]);
            }
        }
        // causal mask + spill scores to LDS
#pragma unroll
        for (int nt = 0; nt < 4; ++nt)
#pragma unroll
            for (int i = 0; i < 8; ++i) {
                const int rl    = i + hA * 8;
                const int qglob = qt * 128 + wv * 16 + rl;
                const int kglob = kb + nt * 16 + rA;
                float v = sc[nt][i];
                if (kglob > qglob) v = -1e9f;
                SL[wv * 16 + rl][nt * 16 + rA] = v;
            }
        __syncthreads();

        // online softmax: one thread per query row
        if (t < 128) {
            const float mo = mL[t];
            float mx = mo;
#pragma unroll 8
            for (int c = 0; c < 64; ++c) mx = fmaxf(mx, SL[t][c]);
            const float alpha = __expf(mo - mx);
            float sum = 0.f;
#pragma unroll 8
            for (int c = 0; c < 64; ++c) {
                const float p = __expf(SL[t][c] - mx);
                PL[t][c] = (bf16)p;
                sum += p;
            }
            mL[t] = mx;
            lL[t] = alpha * lL[t] + sum;
            alphaL[t] = alpha;
        }
        __syncthreads();

        // rescale O by alpha, then O += P @ V
#pragma unroll
        for (int i = 0; i < 8; ++i) {
            const float al = alphaL[wv * 16 + i + hA * 8];
#pragma unroll
            for (int nt = 0; nt < 4; ++nt) O[nt][i] *= al;
        }
#pragma unroll
        for (int kk = 0; kk < 64; kk += 32) {
            FragBF a;
            a.u[0] = *(const u32x4*)&PL[wv * 16 + rA][kk + hA * 8];
            a.u[1] = *(const u32x4*)&PL[wv * 16 + rA][kk + 16 + hA * 8];
#pragma unroll
            for (int nt = 0; nt < 4; ++nt) {
                FragBF b;
                for (int s = 0; s < 8; ++s) {
                    b.v[s]     = VL[kk + hA * 8 + s][nt * 16 + rA];
                    b.v[8 + s] = VL[kk + 16 + hA * 8 + s][nt * 16 + rA];
                }
                O[nt] = wmma_bf16(a.v, b.v, O[nt]);
            }
        }
    }
    __syncthreads();

    // normalize by l and write merged-head layout [B,S,D] bf16
#pragma unroll
    for (int nt = 0; nt < 4; ++nt)
#pragma unroll
        for (int i = 0; i < 8; ++i) {
            const int rl  = i + hA * 8;
            const int row = wv * 16 + rl;
            const float v = O[nt][i] / lL[row];
            const int sg  = qt * 128 + row;
            aws[((size_t)bb * S + sg) * D + h * HD + nt * 16 + rA] = (bf16)v;
        }
}

// ---------------------------------------------------------------------------
// Kernel 3: out = a @ w_proj + b_proj  (M=8192, K=1024, N=1024), fp32 out.
// A tile is bf16 already -> async memory->LDS staging.
// ---------------------------------------------------------------------------
__global__ __launch_bounds__(256) void k_proj(const bf16* __restrict__ a,
                                              const float* __restrict__ w,
                                              const float* __restrict__ bias,
                                              float* __restrict__ out) {
    __shared__ __align__(16) bf16 AL[128][32];
    __shared__ __align__(16) bf16 BL[32][128];

    const int t     = threadIdx.x;
    const int lane  = t & 31;
    const int wv    = t >> 5;
    const int waveM = wv & 3;
    const int waveN = wv >> 2;
    const int m0    = blockIdx.y * 128;
    const int n0    = blockIdx.x * 128;

    v8f acc[2][4];
#pragma unroll
    for (int mt = 0; mt < 2; ++mt)
#pragma unroll
        for (int nt = 0; nt < 4; ++nt)
            acc[mt][nt] = (v8f){0.f,0.f,0.f,0.f,0.f,0.f,0.f,0.f};

    const int rA = lane & 15;
    const int hA = lane >> 4;

    for (int k0 = 0; k0 < D; k0 += 32) {
        {   // A tile: already bf16 -> async straight copy
            const int r  = t >> 1;
            const int cb = (t & 1) * 16;
            const bf16* src = a + (size_t)(m0 + r) * D + k0 + cb;
            async_copy16(src,     &AL[r][cb]);
            async_copy16(src + 8, &AL[r][cb + 8]);
        }
        {   // B tile: f32 -> bf16
            const int r  = t >> 3;
            const int cb = (t & 7) * 16;
            const float4* src = (const float4*)(w + (size_t)(k0 + r) * D + n0 + cb);
            bf16 tmp[16];
#pragma unroll
            for (int j = 0; j < 4; ++j) {
                float4 f = src[j];
                tmp[4*j+0] = (bf16)f.x; tmp[4*j+1] = (bf16)f.y;
                tmp[4*j+2] = (bf16)f.z; tmp[4*j+3] = (bf16)f.w;
            }
            *(u32x4*)&BL[r][cb]     = ((u32x4*)tmp)[0];
            *(u32x4*)&BL[r][cb + 8] = ((u32x4*)tmp)[1];
        }
        wait_async();
        __syncthreads();

        FragBF af[2], bf[4];
#pragma unroll
        for (int mt = 0; mt < 2; ++mt) {
            const int row = waveM * 32 + mt * 16 + rA;
            af[mt].u[0] = *(const u32x4*)&AL[row][hA * 8];
            af[mt].u[1] = *(const u32x4*)&AL[row][16 + hA * 8];
        }
#pragma unroll
        for (int nt = 0; nt < 4; ++nt) {
            const int col = waveN * 64 + nt * 16;
            for (int s = 0; s < 8; ++s) {
                bf[nt].v[s]     = BL[hA * 8 + s][col + rA];
                bf[nt].v[8 + s] = BL[16 + hA * 8 + s][col + rA];
            }
        }
#pragma unroll
        for (int mt = 0; mt < 2; ++mt)
#pragma unroll
            for (int nt = 0; nt < 4; ++nt)
                acc[mt][nt] = wmma_bf16(af[mt].v, bf[nt].v, acc[mt][nt]);
        __syncthreads();
    }

    const int cN   = lane & 15;
    const int hsel = lane >> 4;
#pragma unroll
    for (int mt = 0; mt < 2; ++mt)
#pragma unroll
        for (int nt = 0; nt < 4; ++nt)
#pragma unroll
            for (int i = 0; i < 8; ++i) {
                const int m = m0 + waveM * 32 + mt * 16 + i + hsel * 8;
                const int n = n0 + waveN * 64 + nt * 16 + cN;
                out[(size_t)m * D + n] = acc[mt][nt][i] + bias[n];
            }
}

// ---------------------------------------------------------------------------
// Launch: inputs are (x, w_attn, b_attn, w_proj, b_proj), all fp32.
// Workspace layout (bf16): q[B,H,S,64] | kT[B,H,64,S] | v[B,H,S,64] | a[B,S,D]
// = 16 MB each, 64 MB total.
// ---------------------------------------------------------------------------
extern "C" void kernel_launch(void* const* d_in, const int* in_sizes, int n_in,
                              void* d_out, int out_size, void* d_ws, size_t ws_size,
                              hipStream_t stream) {
    (void)in_sizes; (void)n_in; (void)out_size; (void)ws_size;
    const float* x      = (const float*)d_in[0];
    const float* w_attn = (const float*)d_in[1];
    const float* b_attn = (const float*)d_in[2];
    const float* w_proj = (const float*)d_in[3];
    const float* b_proj = (const float*)d_in[4];
    float* out = (float*)d_out;

    char* ws = (char*)d_ws;
    const size_t SEG = (size_t)16 * 1024 * 1024;
    bf16* qws  = (bf16*)(ws + 0 * SEG);
    bf16* ktws = (bf16*)(ws + 1 * SEG);
    bf16* vws  = (bf16*)(ws + 2 * SEG);
    bf16* aws  = (bf16*)(ws + 3 * SEG);

    k_gemm_qkv<<<dim3(D3 / 128, (Bc * S) / 128), 256, 0, stream>>>(
        x, w_attn, b_attn, qws, ktws, vws);
    k_flash<<<dim3(S / 128, Bc * H), 256, 0, stream>>>(qws, ktws, vws, aws);
    k_proj<<<dim3(D / 128, (Bc * S) / 128), 256, 0, stream>>>(
        aws, w_proj, b_proj, out);
}
